// CausalSelfAttention_300647711154
// MI455X (gfx1250) — matrix operand order repaired and checked by reading, not compiled, not run on hardware
//
#include <hip/hip_runtime.h>
#include <hip/hip_bf16.h>
#include <math.h>
#include <stdint.h>

// ---------------------------------------------------------------------------
// Causal self-attention for MI455X (gfx1250, wave32, WMMA, async-to-LDS with
// double-buffered pipeline).
//   B=4, T=2048, D=1024, H=16, HD=64, M = B*T = 8192.
// All matmuls: v_wmma_f32_16x16x32_bf16 (bf16 operands converted ONCE,
// fp32 accumulate).  x / W pre-converted to bf16; GEMMs emit bf16 copies of
// Q/K/V for the attention stage.  LDS tiles are double-buffered: the next
// tile's global_load_async_to_lds_b128 is in flight while WMMA consumes the
// current tile (s_wait_asynccnt 2 == current tile resident, in-order cnt).
// ---------------------------------------------------------------------------

#define BB 4
#define TT 2048
#define DD 1024
#define HH 16
#define HDD 64
#define MM (BB * TT)

typedef __bf16 bf16;
typedef __attribute__((ext_vector_type(16))) __bf16 v16bf;
typedef __attribute__((ext_vector_type(8)))  __bf16 v8bf;
typedef __attribute__((ext_vector_type(4)))  __bf16 v4bf;
typedef __attribute__((ext_vector_type(8)))  float  v8f;
typedef __attribute__((ext_vector_type(4)))  float  v4f;

__device__ __forceinline__ v8f wmma_bf16(const v16bf& a, const v16bf& b, const v8f& c) {
    return __builtin_amdgcn_wmma_f32_16x16x32_bf16(
        false, a, false, b, (short)0, c, false, false);
}

__device__ __forceinline__ v16bf cat16(const v8bf& lo, const v8bf& hi) {
    return __builtin_shufflevector(lo, hi, 0,1,2,3,4,5,6,7,8,9,10,11,12,13,14,15);
}

// 16B async copy global -> LDS (ASYNCcnt path, §15.18 op 98)
__device__ __forceinline__ void async_cp16(const bf16* g, bf16* l) {
    uint32_t lo = (uint32_t)(uintptr_t)l;   // low 32 bits of generic = LDS offset
    asm volatile("global_load_async_to_lds_b128 %0, %1, off"
                 :: "v"(lo), "v"(g) : "memory");
}
// wait: current tile resident (2 newest chunks of the next tile may remain)
__device__ __forceinline__ void async_wait2() {
    asm volatile("s_wait_asynccnt 0x2" ::: "memory");
}
__device__ __forceinline__ void async_wait0() {
    asm volatile("s_wait_asynccnt 0x0" ::: "memory");
}

// ---------------------------------------------------------------------------
// fp32 -> bf16 elementwise (one-time conversion of x and weights)
// ---------------------------------------------------------------------------
__global__ __launch_bounds__(256)
void cvt_f32_bf16(const float* __restrict__ src, bf16* __restrict__ dst, int n4)
{
    int i = blockIdx.x * 256 + threadIdx.x;
    if (i < n4) {
        v4f f = *(const v4f*)(src + (size_t)i * 4);
        v4bf o;
        #pragma unroll
        for (int j = 0; j < 4; ++j) o[j] = (bf16)f[j];
        *(v4bf*)(dst + (size_t)i * 4) = o;
    }
}

// ---------------------------------------------------------------------------
// GEMM: out[m,n] = sum_k A[m,k]*W[n,k] + bias[n]   (A @ W^T + b), bf16 inputs.
// Block = 256 thr = 8 waves.  Block tile 64(M) x 128(N); wave = 64x16 =
// 4 WMMA M-subtiles reusing one B fragment.  Double-buffered async x tile.
// headMode=1: scatter (h,hd) -> [B,H,T,64]; else flat [M,D].
// ---------------------------------------------------------------------------
#define KCHUNK 64
#define XPITCH 72   // bf16 units; 144B row pitch (16B aligned, staggered)

__global__ __launch_bounds__(256)
void gemm_bf16_wmma(const bf16* __restrict__ A, const bf16* __restrict__ W,
                    const float* __restrict__ bias,
                    float* __restrict__ outF, bf16* __restrict__ outB,
                    int headMode)
{
    __shared__ __align__(16) bf16 sx[2][64 * XPITCH];

    const int tid   = threadIdx.x;
    const int lane  = tid & 31;
    const int wave  = tid >> 5;
    const int laneN = lane & 15;
    const int laneH = lane >> 4;

    const int mBase = blockIdx.y * 64;
    const int n     = blockIdx.x * 128 + wave * 16 + laneN;
    const bf16* __restrict__ Wrow = W + (size_t)n * DD;

    // 64 rows x 64 bf16 = 512 x 16B chunks, 2 async chunks per thread/wave
    auto stage = [&](int k0, int buf) {
        #pragma unroll
        for (int c = 0; c < 2; ++c) {
            const int idx = tid + c * 256;          // 0..511
            const int row = idx >> 3;
            const int c8  = (idx & 7) * 8;
            async_cp16(A + (size_t)(mBase + row) * DD + k0 + c8,
                       &sx[buf][row * XPITCH + c8]);
        }
    };

    v8f acc[4] = {{}, {}, {}, {}};

    stage(0, 0);
    int ib = 0;
    for (int k0 = 0; k0 < DD; k0 += KCHUNK, ib ^= 1) {
        const bool more = (k0 + KCHUNK) < DD;
        if (more) {
            stage(k0 + KCHUNK, ib ^ 1);             // overlap next DMA with math
            __builtin_prefetch(Wrow + k0 + KCHUNK, 0, 3);
            async_wait2();
        } else {
            async_wait0();
        }
        __syncthreads();

        #pragma unroll
        for (int kk = 0; kk < KCHUNK; kk += 32) {
            // B fragment (32x16): lane = column n, k = {kk, kk+16} + laneH*8 + [0..7]
            const bf16* wp = Wrow + k0 + kk + laneH * 8;
            v8bf b0 = *(const v8bf*)(wp);
            v8bf b1 = *(const v8bf*)(wp + 16);
            const v16bf bfrag = cat16(b0, b1);

            #pragma unroll
            for (int ms = 0; ms < 4; ++ms) {
                // A fragment (16x32): lane row = ms*16+laneN,
                // k = {kk, kk+16} + laneH*8 + [0..7]
                const bf16* ap = &sx[ib][(ms * 16 + laneN) * XPITCH + kk + laneH * 8];
                v8bf a0 = *(const v8bf*)(ap);
                v8bf a1 = *(const v8bf*)(ap + 16);
                acc[ms] = wmma_bf16(cat16(a0, a1), bfrag, acc[ms]);
            }
        }
        __syncthreads();                            // free sx[ib] for restaging
    }

    const float bval = bias[n];
    if (headMode) {
        const int h = n >> 6, hd = n & 63;
        #pragma unroll
        for (int ms = 0; ms < 4; ++ms)
        #pragma unroll
        for (int v = 0; v < 8; ++v) {
            const float r = acc[ms][v] + bval;
            const int m = mBase + ms * 16 + laneH * 8 + v;
            const int b = m >> 11, t = m & (TT - 1);
            const size_t idx = (((size_t)b * HH + h) * TT + t) * HDD + hd;
            if (outF) outF[idx] = r;
            if (outB) outB[idx] = (bf16)r;
        }
    } else {
        #pragma unroll
        for (int ms = 0; ms < 4; ++ms)
        #pragma unroll
        for (int v = 0; v < 8; ++v) {
            const int m = mBase + ms * 16 + laneH * 8 + v;
            outF[(size_t)m * DD + n] = acc[ms][v] + bval;
        }
    }
}

// ---------------------------------------------------------------------------
// Flash attention (causal).  Q,K,V bf16 [B*H,T,64]; output bf16 [B,T,D]
// (A-operand layout for the final projection).  Block = 128 thr = 4 waves,
// wave = 16 queries.  Double-buffered K (async-to-LDS) and V^T (transposed
// in LDS so PV B-fragments are contiguous ds reads).  Uniform loop over the
// block's key range; per-wave compute guard is wave-uniform (EXEC all-ones).
// ---------------------------------------------------------------------------
__global__ __launch_bounds__(128)
void flash_attn_wmma(const bf16* __restrict__ Q, const bf16* __restrict__ K,
                     const bf16* __restrict__ V, bf16* __restrict__ Ob)
{
    __shared__ __align__(16) bf16 sk[2][32 * 72];   // K tile [key][hd], pitch 72
    __shared__ __align__(16) bf16 svt[2][64 * 48];  // V^T    [hd][key], pitch 48
    __shared__ __align__(16) bf16 spt[4][16 * 48];  // per-wave P [row][key]

    const int tid   = threadIdx.x;
    const int lane  = tid & 31;
    const int wave  = tid >> 5;
    const int laneN = lane & 15;
    const int laneH = lane >> 4;

    const int bh    = blockIdx.y;                   // 0..63
    const int qBase = blockIdx.x * 64 + wave * 16;
    const int qEnd  = qBase + 16;
    const int qMax  = blockIdx.x * 64 + 64;

    const bf16* __restrict__ Qb = Q + (size_t)bh * TT * HDD;
    const bf16* __restrict__ Kb = K + (size_t)bh * TT * HDD;
    const bf16* __restrict__ Vb = V + (size_t)bh * TT * HDD;

    // stage one 32-key tile: K async (2 chunks/wave) + V transposed via DS
    auto stageKV = [&](int kb, int buf) {
        #pragma unroll
        for (int c = 0; c < 2; ++c) {
            const int idx = tid + c * 128;          // 0..255
            const int row = idx >> 3;
            const int c8  = (idx & 7) * 8;
            async_cp16(Kb + (size_t)(kb + row) * HDD + c8,
                       &sk[buf][row * 72 + c8]);
        }
        const int key = tid >> 2;
        const int hd0 = (tid & 3) * 16;
        const bf16* vr = Vb + (size_t)(kb + key) * HDD + hd0;
        v8bf v0 = *(const v8bf*)(vr);
        v8bf v1 = *(const v8bf*)(vr + 8);
        #pragma unroll
        for (int j = 0; j < 8; ++j) {
            svt[buf][(hd0 + j) * 48 + key]     = v0[j];
            svt[buf][(hd0 + 8 + j) * 48 + key] = v1[j];
        }
    };

    // Q A-fragments (row = qBase + laneN), k-chunks 0 / 32
    v16bf qf[2];
    {
        const bf16* qrow = Qb + (size_t)(qBase + laneN) * HDD;
        #pragma unroll
        for (int c = 0; c < 2; ++c) {
            v8bf q0 = *(const v8bf*)(qrow + c * 32 + laneH * 8);
            v8bf q1 = *(const v8bf*)(qrow + c * 32 + laneH * 8 + 16);
            qf[c] = cat16(q0, q1);
        }
    }

    v8f oacc[4] = {{}, {}, {}, {}};
    float mrow[8], lrow[8];
    #pragma unroll
    for (int v = 0; v < 8; ++v) { mrow[v] = -INFINITY; lrow[v] = 0.0f; }

    const float scale = 0.125f;   // 1/sqrt(64)

    stageKV(0, 0);
    int ib = 0;
    for (int kb = 0; kb < qMax; kb += 32, ib ^= 1) {
        const bool more = (kb + 32) < qMax;
        if (more) { stageKV(kb + 32, ib ^ 1); async_wait2(); }
        else      { async_wait0(); }
        __syncthreads();

        if (kb < qEnd) {   // wave-uniform causal skip
            // ---- S = Q K^T : K B-fragments from LDS
            v8f s01[2];
            #pragma unroll
            for (int s = 0; s < 2; ++s) {
                v8f sc = {};
                #pragma unroll
                for (int c = 0; c < 2; ++c) {
                    const bf16* kp = &sk[ib][(s * 16 + laneN) * 72 + c * 32 + laneH * 8];
                    v8bf k0 = *(const v8bf*)(kp);
                    v8bf k1 = *(const v8bf*)(kp + 16);
                    sc = wmma_bf16(qf[c], cat16(k0, k1), sc);
                }
                s01[s] = sc;
            }

            // ---- online softmax (row = v + 8*laneH, col = kb + s*16 + laneN)
            #pragma unroll
            for (int v = 0; v < 8; ++v) {
                const int qi  = qBase + laneH * 8 + v;
                const int k0i = kb + laneN;
                const int k1i = kb + 16 + laneN;
                float e0 = (k0i <= qi) ? s01[0][v] * scale : -INFINITY;
                float e1 = (k1i <= qi) ? s01[1][v] * scale : -INFINITY;

                float mx = fmaxf(e0, e1);
                #pragma unroll
                for (int off = 1; off < 16; off <<= 1)
                    mx = fmaxf(mx, __shfl_xor(mx, off, 32));

                const float mnew  = fmaxf(mrow[v], mx);
                const float alpha = __expf(mrow[v] - mnew);
                const float p0 = __expf(e0 - mnew);
                const float p1 = __expf(e1 - mnew);

                float rs = p0 + p1;
                #pragma unroll
                for (int off = 1; off < 16; off <<= 1)
                    rs += __shfl_xor(rs, off, 32);

                lrow[v] = lrow[v] * alpha + rs;
                mrow[v] = mnew;
                #pragma unroll
                for (int c = 0; c < 4; ++c) oacc[c][v] *= alpha;

                const int r = laneH * 8 + v;
                spt[wave][r * 48 + laneN]      = (bf16)p0;
                spt[wave][r * 48 + 16 + laneN] = (bf16)p1;
            }

            // per-wave LDS RAW: DS ops are in-order, make the wait explicit
            asm volatile("s_wait_dscnt 0x0" ::: "memory");

            // ---- P as A-fragment (16x32)
            const bf16* pr = &spt[wave][laneN * 48 + laneH * 8];
            v8bf p0 = *(const v8bf*)(pr);
            v8bf p1 = *(const v8bf*)(pr + 16);
            const v16bf pf = cat16(p0, p1);

            // ---- PV: V^T B-fragments from LDS (keys contiguous)
            #pragma unroll
            for (int c = 0; c < 4; ++c) {
                const bf16* vp = &svt[ib][(c * 16 + laneN) * 48 + laneH * 8];
                v8bf vv0 = *(const v8bf*)(vp);
                v8bf vv1 = *(const v8bf*)(vp + 16);
                oacc[c] = wmma_bf16(pf, cat16(vv0, vv1), oacc[c]);
            }
        }
        __syncthreads();    // free sk/svt[ib] for restaging
    }

    // ---- epilogue: normalize, emit bf16 in [B,T,D] (A layout for out-proj)
    const int b = bh >> 4, h = bh & 15;
    #pragma unroll
    for (int v = 0; v < 8; ++v) {
        const float inv = 1.0f / lrow[v];
        const int t = qBase + laneH * 8 + v;
        bf16* orow = Ob + ((size_t)b * TT + t) * DD + h * HDD;
        #pragma unroll
        for (int c = 0; c < 4; ++c)
            orow[c * 16 + laneN] = (bf16)(oacc[c][v] * inv);
    }
}

// ---------------------------------------------------------------------------
// d_out = [ out (8M f32) | k (8M f32) | v (8M f32) ]
// d_ws  = [ xbf 16MB | Wq..Wo bf16 4x2MB | qbf 16MB | kbf 16MB | vbf 16MB |
//           aobf 16MB ]  (= 88MB)
// ---------------------------------------------------------------------------
extern "C" void kernel_launch(void* const* d_in, const int* in_sizes, int n_in,
                              void* d_out, int out_size, void* d_ws, size_t ws_size,
                              hipStream_t stream)
{
    const float* x  = (const float*)d_in[0];
    const float* Wq = (const float*)d_in[1];
    const float* bq = (const float*)d_in[2];
    const float* Wk = (const float*)d_in[3];
    const float* bk = (const float*)d_in[4];
    const float* Wv = (const float*)d_in[5];
    const float* bv = (const float*)d_in[6];
    const float* Wo = (const float*)d_in[7];
    const float* bo = (const float*)d_in[8];

    float* outp = (float*)d_out;                        // [B,T,D]
    float* kout = outp + (size_t)MM * DD;               // [B,H,T,64]
    float* vout = kout + (size_t)MM * DD;               // [B,H,T,64]

    bf16* xbf  = (bf16*)d_ws;                           // [M, D]
    bf16* wqb  = xbf + (size_t)MM * DD;                 // [D, D] each
    bf16* wkb  = wqb + (size_t)DD * DD;
    bf16* wvb  = wkb + (size_t)DD * DD;
    bf16* wob  = wvb + (size_t)DD * DD;
    bf16* qbf  = wob + (size_t)DD * DD;                 // [B*H, T, 64]
    bf16* kbf  = qbf + (size_t)MM * DD;
    bf16* vbf  = kbf + (size_t)MM * DD;
    bf16* aobf = vbf + (size_t)MM * DD;                 // [M, D]

    const int nX4 = MM * DD / 4;       // 2097152
    const int nW4 = DD * DD / 4;       // 262144
    cvt_f32_bf16<<<nX4 / 256, 256, 0, stream>>>(x,  xbf, nX4);
    cvt_f32_bf16<<<nW4 / 256, 256, 0, stream>>>(Wq, wqb, nW4);
    cvt_f32_bf16<<<nW4 / 256, 256, 0, stream>>>(Wk, wkb, nW4);
    cvt_f32_bf16<<<nW4 / 256, 256, 0, stream>>>(Wv, wvb, nW4);
    cvt_f32_bf16<<<nW4 / 256, 256, 0, stream>>>(Wo, wob, nW4);

    dim3 gGemm(DD / 128, MM / 64);     // (8, 128)
    gemm_bf16_wmma<<<gGemm, dim3(256), 0, stream>>>(xbf, wqb, bq, nullptr, qbf, 1);
    gemm_bf16_wmma<<<gGemm, dim3(256), 0, stream>>>(xbf, wkb, bk, kout,    kbf, 1);
    gemm_bf16_wmma<<<gGemm, dim3(256), 0, stream>>>(xbf, wvb, bv, vout,    vbf, 1);

    dim3 gAttn(TT / 64, BB * HH);      // (32, 64)
    flash_attn_wmma<<<gAttn, dim3(128), 0, stream>>>(qbf, kbf, vbf, aobf);

    gemm_bf16_wmma<<<gGemm, dim3(256), 0, stream>>>(aobf, wob, bo, outp, nullptr, 0);
}
